// RecurrentEmbeddingLayer_75728863363318
// MI455X (gfx1250) — matrix operand order repaired and checked by reading, not compiled, not run on hardware
//
#include <hip/hip_runtime.h>
#include <stdint.h>

// Problem constants
#define BB 128
#define TT 1024
#define FF 16
#define H1 128
#define H2 128
#define EE 64
#define MROWS (BB * TT)   // 131072 flattened (b,t) rows

typedef __attribute__((ext_vector_type(16))) __bf16 v16bf;
typedef __attribute__((ext_vector_type(8)))  float  v8f;

__device__ __forceinline__ unsigned short f2bf(float f) {
    union { float f; uint32_t u; } c; c.f = f;
    uint32_t r = (c.u + 0x7FFFu + ((c.u >> 16) & 1u)) >> 16;
    return (unsigned short)r;
}

// ---- WMMA helper: D = A(16x32 bf16) * B(32x16 bf16) + C (f32) ----
__device__ __forceinline__ v8f wmma_bf16(const uint32_t aw[8], const uint32_t bw[8], v8f c) {
    union { v16bf v; uint32_t w[8]; } A, Bm;
#pragma unroll
    for (int v = 0; v < 8; ++v) { A.w[v] = aw[v]; Bm.w[v] = bw[v]; }
    return __builtin_amdgcn_wmma_f32_16x16x32_bf16(false, A.v, false, Bm.v,
                                                   (short)0, c, false, false);
}

// A fragment (16x32, row-major source, stride ld): lanes 0-15 -> M rows, half selects K sub-block.
// VGPR v<4: K = half*8 + 2v ; v>=4: K = 16 + half*8 + 2(v-4). 2 consecutive bf16 per dword.
__device__ __forceinline__ void load_a(const unsigned short* p, int ld, int lane, uint32_t aw[8]) {
    const unsigned short* pr = p + (lane & 15) * ld;
    int half8 = (lane >> 4) * 8;
#pragma unroll
    for (int v = 0; v < 8; ++v) {
        int k = (v < 4) ? (half8 + 2 * v) : (16 + half8 + 2 * (v - 4));
        aw[v] = *(const uint32_t*)(pr + k);
    }
}

// B fragment (32x16) from N-major (transposed) source: row n holds K contiguous.
// lane: N = lane&15 ; v<4: K = half*8 + 2v ; v>=4: K = 16 + half*8 + 2(v-4).
__device__ __forceinline__ void load_b(const unsigned short* p, int ld, int lane, uint32_t bw[8]) {
    const unsigned short* pr = p + (lane & 15) * ld;
    int half8 = (lane >> 4) * 8;
#pragma unroll
    for (int v = 0; v < 8; ++v)
        bw[v] = *(const uint32_t*)(pr + ((v < 4) ? (half8 + 2 * v) : (16 + half8 + 2 * (v - 4))));
}

// ---------- prep kernels ----------
// batch int32 (B*T,16) -> bf16 A matrix (B*T,32) zero-padded in K
__global__ void prep_xa0(const int* __restrict__ batch, unsigned short* __restrict__ xa) {
    int idx = blockIdx.x * 256 + threadIdx.x;          // over MROWS*32
    int bt = idx >> 5, k = idx & 31;
    xa[idx] = (k < FF) ? f2bf((float)batch[bt * FF + k]) : (unsigned short)0;
}

// kernel f32 (K,N) -> transposed bf16 (N,Kp), zero-padded K
__global__ void conv_kt(const float* __restrict__ src, unsigned short* __restrict__ dst,
                        int K, int N, int Kp) {
    int idx = blockIdx.x * 256 + threadIdx.x;
    if (idx >= N * Kp) return;
    int n = idx / Kp, k = idx % Kp;
    dst[idx] = (k < K) ? f2bf(src[k * N + n]) : (unsigned short)0;
}

// ---------- xproj GEMM: out(M,N) = A(M,Kp) * Bt(N,Kp)^T + bias ----------
__global__ void xproj_gemm(const unsigned short* __restrict__ A, int Kp,
                           const unsigned short* __restrict__ Bt,
                           const float* __restrict__ bias,
                           float* __restrict__ out, int N) {
    int tilesN = N >> 4;
    int waveId = blockIdx.x * 8 + (threadIdx.x >> 5);
    int lane = threadIdx.x & 31;
    int tm = waveId / tilesN, tn = waveId % tilesN;
    int col = tn * 16 + (lane & 15);
    int half = lane >> 4;
    float bv = bias[col];
    v8f acc;
#pragma unroll
    for (int j = 0; j < 8; ++j) acc[j] = bv;
    for (int kt = 0; kt < Kp; kt += 32) {
        uint32_t aw[8], bw[8];
        load_a(A + (long)(tm * 16) * Kp + kt, Kp, lane, aw);
        load_b(Bt + (long)(tn * 16) * Kp + kt, Kp, lane, bw);
        acc = wmma_bf16(aw, bw, acc);
    }
#pragma unroll
    for (int j = 0; j < 8; ++j)
        out[(long)(tm * 16 + j + half * 8) * N + col] = acc[j];
}

// ---------- persistent GRU scan: 1 WG per 16 batch rows, loop over T ----------
template <int U, bool LAST>
__global__ void gru_scan(const float* __restrict__ xproj, const float* __restrict__ rk,
                         const float* __restrict__ bias, const int* __restrict__ batch,
                         unsigned short* __restrict__ outb,
                         float* __restrict__ outf, float* __restrict__ statef) {
    constexpr int N3 = 3 * U;
    constexpr int NT = N3 / 16;   // output tiles per step
    constexpr int KT = U / 32;    // K slabs per tile
    extern __shared__ char smem[];
    unsigned short* rkT = (unsigned short*)smem;            // (N3, U) bf16
    float* brec = (float*)(rkT + N3 * U);                   // N3
    float* rec  = brec + N3;                                // 16 x N3
    float* hf   = rec + 16 * N3;                            // 16 x U
    float* op   = hf + 16 * U;                              // 16 x U
    unsigned short* h16 = (unsigned short*)(op + 16 * U);   // 16 x U
    int* maskS  = (int*)(h16 + 16 * U);                     // 16

    const int tid = threadIdx.x;
    const int r0 = blockIdx.x * 16;

    for (int i = tid; i < N3 * U; i += 256) {
        int n = i / U, k = i % U;
        rkT[i] = f2bf(rk[k * N3 + n]);
    }
    for (int i = tid; i < N3; i += 256) brec[i] = bias[N3 + i];
    for (int i = tid; i < 16 * U; i += 256) { hf[i] = 0.f; op[i] = 0.f; h16[i] = 0; }
    __syncthreads();

    const int wave = tid >> 5, lane = tid & 31;
    const int colIn = lane & 15, half = lane >> 4;

    for (int t = 0; t < TT; ++t) {
        if (tid < 16)
            maskS[tid] = (batch[(((long)(r0 + tid)) * TT + t) * FF + (FF - 1)] != -1) ? 1 : 0;
        // rec = h @ rk + b_rec  (bf16 WMMA, f32 accumulate)
        for (int n = wave; n < NT; n += 8) {
            float bv = brec[n * 16 + colIn];
            v8f acc;
#pragma unroll
            for (int j = 0; j < 8; ++j) acc[j] = bv;
#pragma unroll
            for (int kt = 0; kt < KT; ++kt) {
                uint32_t aw[8], bw[8];
                load_a(h16 + kt * 32, U, lane, aw);
                load_b(rkT + (n * 16) * U + kt * 32, U, lane, bw);
                acc = wmma_bf16(aw, bw, acc);
            }
#pragma unroll
            for (int j = 0; j < 8; ++j)
                rec[(j + half * 8) * N3 + n * 16 + colIn] = acc[j];
        }
        __syncthreads();
        // gates (Keras reset_after: r multiplies recurrent h-part only)
        for (int i = tid; i < 16 * U; i += 256) {
            int m = i / U, u = i % U;
            long bt = ((long)(r0 + m)) * TT + t;
            const float* xp = xproj + bt * N3;
            float rz = rec[m * N3 + u];
            float rr = rec[m * N3 + U + u];
            float rh = rec[m * N3 + 2 * U + u];
            float z  = 1.f / (1.f + expf(-(xp[u] + rz)));
            float r  = 1.f / (1.f + expf(-(xp[U + u] + rr)));
            float hh = tanhf(xp[2 * U + u] + r * rh);
            float hold = hf[i];
            float hnew = z * hold + (1.f - z) * hh;
            bool mk = maskS[m] != 0;
            float hn = mk ? hnew : hold;
            float o  = mk ? hnew : op[i];
            hf[i] = hn; op[i] = o; h16[i] = f2bf(hn);
            if (LAST) outf[bt * U + u] = o;
            else      outb[bt * U + u] = f2bf(o);
            if (t + 1 < TT) __builtin_prefetch(xp + N3, 0, 0);  // next-step xproj
        }
        __syncthreads();
    }
    if (LAST) {
        for (int i = tid; i < 16 * U; i += 256) {
            int m = i / U, u = i % U;
            statef[(r0 + m) * U + u] = hf[i];
        }
    }
}

static inline size_t alignUp(size_t x) { return (x + 255) & ~size_t(255); }

extern "C" void kernel_launch(void* const* d_in, const int* in_sizes, int n_in,
                              void* d_out, int out_size, void* d_ws, size_t ws_size,
                              hipStream_t stream) {
    (void)in_sizes; (void)n_in; (void)out_size; (void)ws_size;
    const int*   batch = (const int*)d_in[0];
    const float* k0 = (const float*)d_in[1];
    const float* rk0 = (const float*)d_in[2];
    const float* b0 = (const float*)d_in[3];
    const float* k1 = (const float*)d_in[4];
    const float* rk1 = (const float*)d_in[5];
    const float* b1 = (const float*)d_in[6];
    const float* k2 = (const float*)d_in[7];
    const float* rk2 = (const float*)d_in[8];
    const float* b2 = (const float*)d_in[9];

    float* outf   = (float*)d_out;                 // (B,T,E)
    float* statef = outf + (size_t)MROWS * EE;     // (B,E)

    // workspace carving
    char* p = (char*)d_ws; size_t off = 0;
    unsigned short* xa0 = (unsigned short*)(p + off); off = alignUp(off + (size_t)MROWS * 32 * 2);
    unsigned short* kt0 = (unsigned short*)(p + off); off = alignUp(off + (size_t)384 * 32 * 2);
    unsigned short* kt1 = (unsigned short*)(p + off); off = alignUp(off + (size_t)384 * 128 * 2);
    unsigned short* kt2 = (unsigned short*)(p + off); off = alignUp(off + (size_t)192 * 128 * 2);
    float* xproj        = (float*)(p + off);        off = alignUp(off + (size_t)MROWS * 384 * 4);
    unsigned short* oA  = (unsigned short*)(p + off); off = alignUp(off + (size_t)MROWS * 128 * 2);
    unsigned short* oB  = (unsigned short*)(p + off); off = alignUp(off + (size_t)MROWS * 128 * 2);

    // shared-mem sizes for the scan kernels
    auto smemFor = [](int U) -> size_t {
        int N3 = 3 * U;
        return (size_t)N3 * U * 2 + N3 * 4 + 16 * N3 * 4 + 16 * U * 4 * 2 + 16 * U * 2 + 64;
    };

    // prep
    prep_xa0<<<(MROWS * 32) / 256, 256, 0, stream>>>(batch, xa0);
    conv_kt<<<(384 * 32 + 255) / 256, 256, 0, stream>>>(k0, kt0, FF, 3 * H1, 32);
    conv_kt<<<(384 * 128 + 255) / 256, 256, 0, stream>>>(k1, kt1, H1, 3 * H2, 128);
    conv_kt<<<(192 * 128 + 255) / 256, 256, 0, stream>>>(k2, kt2, H2, 3 * EE, 128);

    // layer 0
    xproj_gemm<<<(MROWS / 16) * (384 / 16) / 8, 256, 0, stream>>>(xa0, 32, kt0, b0, xproj, 384);
    gru_scan<H1, false><<<BB / 16, 256, smemFor(H1), stream>>>(xproj, rk0, b0, batch, oA, nullptr, nullptr);
    // layer 1
    xproj_gemm<<<(MROWS / 16) * (384 / 16) / 8, 256, 0, stream>>>(oA, 128, kt1, b1, xproj, 384);
    gru_scan<H2, false><<<BB / 16, 256, smemFor(H2), stream>>>(xproj, rk1, b1, batch, oB, nullptr, nullptr);
    // layer 2 (writes d_out directly)
    xproj_gemm<<<(MROWS / 16) * (192 / 16) / 8, 256, 0, stream>>>(oB, 128, kt2, b2, xproj, 192);
    gru_scan<EE, true><<<BB / 16, 256, smemFor(EE), stream>>>(xproj, rk2, b2, batch, nullptr, outf, statef);
}
